// Attention_16612933501287
// MI455X (gfx1250) — matrix operand order repaired and checked by reading, not compiled, not run on hardware
//
#include <hip/hip_runtime.h>

// ---------- types ----------
typedef __attribute__((ext_vector_type(16))) __bf16 v16bf;
typedef __attribute__((ext_vector_type(8)))  float  v8f;

// Problem constants (reference: B=8,N=8,hidden=512,H=W=32,nh=8,hs=64)
#define HIDDEN 512
#define NTOK   8
#define NHEAD  8
#define WTILE  16

// ---------- bf16 helpers (RNE) ----------
static __device__ __forceinline__ __bf16 f2bf(float f) {
  unsigned u = __builtin_bit_cast(unsigned, f);
  unsigned r = (u + 0x7FFFu + ((u >> 16) & 1u)) >> 16;
  unsigned short s = (unsigned short)r;
  return __builtin_bit_cast(__bf16, s);
}
static __device__ __forceinline__ float bf2f(__bf16 h) {
  unsigned short s = __builtin_bit_cast(unsigned short, h);
  unsigned u = (unsigned)s << 16;
  return __builtin_bit_cast(float, u);
}
static __device__ __forceinline__ unsigned rne_bits(float f) {
  unsigned u = __builtin_bit_cast(unsigned, f);
  return u + 0x7FFFu + ((u >> 16) & 1u);
}

// A-fragment (16x32 bf16, wave32) halfword index for element (m=0..15, k=0..31).
// ISA 7.12.2: lanes 0-15 hold K=0-7 (V0-3) and K=16-23 (V4-7); lanes 16-31 hold
// K=8-15 and K=24-31; two bf16 per VGPR, ascending.
static __device__ __forceinline__ int afrag_idx(int m, int k) {
  int lane = m + (((k >> 3) & 1) << 4);
  int e = ((k & 16) >> 1) | (k & 7);
  return lane * 16 + e;
}
// B-fragment (32x16 bf16, wave32) halfword index for element (k=0..31, n=0..15).
// Lanes 0-15: K=0-7,16-23 (N=lane); lanes 16-31: K=8-15,24-31 (N=lane-16).
static __device__ __forceinline__ int bfrag_idx(int k, int n) {
  int lane = n + (((k >> 3) & 1) << 4);
  return lane * 16 + (((k & 16) >> 1) | (k & 7));
}

// ---------------------------------------------------------------------------
// Kernel 1: qv = emb[q[b]] @ Wq + bq ; u[b,ch,head] = g[ch]*hs^-0.5 *
//           sum_d Wk[ch, head*64+d] * qv[head*64+d]   (Wk = Wkv[:, :512])
// ---------------------------------------------------------------------------
__global__ __launch_bounds__(256) void qu_kernel(
    const int* __restrict__ q, const float* __restrict__ emb,
    const float* __restrict__ Wq, const float* __restrict__ bq,
    const float* __restrict__ Wkv, const float* __restrict__ g,
    float* __restrict__ u_g) {
  __shared__ float e_lds[HIDDEN];
  __shared__ float qv_lds[HIDDEN];
  const int t = threadIdx.x, b = blockIdx.x;
  const int qi = q[b];
  for (int i = t; i < HIDDEN; i += 256) e_lds[i] = emb[qi * HIDDEN + i];
  __syncthreads();
  for (int d = t; d < HIDDEN; d += 256) {
    float acc = bq[d];
    for (int ch = 0; ch < HIDDEN; ++ch) acc += e_lds[ch] * Wq[ch * HIDDEN + d];
    qv_lds[d] = acc;
  }
  __syncthreads();
  for (int i = t; i < HIDDEN * NHEAD; i += 256) {
    int ch = i >> 3, head = i & 7;
    const float* wk = Wkv + ch * (2 * HIDDEN) + head * 64;
    const float* qh = qv_lds + head * 64;
    float acc = 0.f;
    #pragma unroll 8
    for (int d = 0; d < 64; ++d) acc += wk[d] * qh[d];
    u_g[b * (HIDDEN * NHEAD) + ch * NHEAD + head] = acc * 0.125f * g[ch];
  }
}

// ---------------------------------------------------------------------------
// Kernel 2: repack g*Wv (=g[ch]*Wkv[ch,512+col]) and Wout to bf16 in
// WMMA B-fragment-ready order: [col_tile][k_chunk][lane][16 halfwords]
// ---------------------------------------------------------------------------
__global__ __launch_bounds__(256) void repack_kernel(
    const float* __restrict__ Wkv, const float* __restrict__ Wout,
    const float* __restrict__ g,
    __bf16* __restrict__ wv_frag, __bf16* __restrict__ wout_frag) {
  int el = blockIdx.x * 256 + threadIdx.x;          // 0..262143
  int ch = el >> 9, col = el & 511;
  int ctg = col >> 4, n = col & 15, chunk = ch >> 5, k = ch & 31;
  int idx = (ctg * 16 + chunk) * 512 + bfrag_idx(k, n);
  if (blockIdx.y == 0) {
    wv_frag[idx] = f2bf(g[ch] * Wkv[ch * (2 * HIDDEN) + HIDDEN + col]);
  } else {
    wout_frag[idx] = f2bf(Wout[ch * HIDDEN + col]);
  }
}

// ---------------------------------------------------------------------------
// Kernel 3: rms[b,n,h,w] = rsqrt(mean_ch c^2 + eps). One wave per (b,n,h),
// lane = w -> contiguous 128B row loads per channel.
// ---------------------------------------------------------------------------
__global__ __launch_bounds__(256) void rms_kernel(const float* __restrict__ c,
                                                  float* __restrict__ rms_g) {
  int gw = blockIdx.x * 8 + (threadIdx.x >> 5);     // 0..2047 = (b,n,h)
  int lane = threadIdx.x & 31;
  int b = gw >> 8, n = (gw >> 5) & 7, h = gw & 31;
  const float* base = c + (((b * NTOK + n) * HIDDEN) * 32 + h) * 32 + lane;
  float acc = 0.f;
  for (int ch = 0; ch < HIDDEN; ++ch) {
    float x = base[ch * 1024];
    acc += x * x;
  }
  rms_g[gw * 32 + lane] = rsqrtf(acc * (1.0f / HIDDEN) + 1e-6f);
}

// ---------------------------------------------------------------------------
// Kernel 4: fused dots -> softmax -> token-mix -> (mixed @ gWv) -> (@ Wout)+b
// One workgroup (8 wave32) per (b, h, 16-w tile).
// ---------------------------------------------------------------------------
__global__ __launch_bounds__(256, 1) void attn_main(
    const float* __restrict__ c, const float* __restrict__ u_g,
    const float* __restrict__ rms_g,
    const __bf16* __restrict__ wv_frag, const __bf16* __restrict__ wout_frag,
    const float* __restrict__ bout, float* __restrict__ out) {
  // A-fragment-ready LDS: [token][k_chunk][512 halfwords]
  __shared__ __bf16 A_frag[NTOK * 16 * 512];        // 128 KB
  __shared__ __bf16 mixed_frag[2 * 16 * 512];       // 32 KB (2 heads in flight)
  __shared__ __bf16 o_frag[16 * 512];               // 16 KB (o tile, A-ready)
  __shared__ float  u_lds[HIDDEN * NHEAD];          // 16 KB
  __shared__ float  rms_s[NTOK * WTILE];            // 512 B
  __shared__ float  dots_lds[WTILE * NHEAD * NTOK]; // 4 KB
  __shared__ float  attn_lds[WTILE * NHEAD * NTOK]; // 4 KB

  const int t = threadIdx.x;
  const int lane = t & 31, wave = t >> 5;
  const int w0 = blockIdx.x * WTILE;
  const int h = blockIdx.y;
  const int b = blockIdx.z;

  // ---- stage u (includes g and hs^-0.5) and rms slice ----
  for (int i = t; i < HIDDEN * NHEAD; i += 256) u_lds[i] = u_g[b * HIDDEN * NHEAD + i];
  if (t < 128) {
    int n = t >> 4, w = t & 15;
    rms_s[n * WTILE + w] = rms_g[((b * NTOK + n) * 32 + h) * 32 + w0 + w];
  }
  __syncthreads();

  // ---- stage A = (c * rms) as bf16, fragment-ready ----
  for (int j = 0; j < 16; ++j) {
    int row = t + 256 * j;                 // (n, ch) row, 64B contiguous load
    int n = row >> 9, ch = row & 511;
    const float* src = c + (((b * NTOK + n) * HIDDEN + ch) * 32 + h) * 32 + w0;
    float4 v0 = *(const float4*)(src + 0);
    float4 v1 = *(const float4*)(src + 4);
    float4 v2 = *(const float4*)(src + 8);
    float4 v3 = *(const float4*)(src + 12);
    float vals[16] = {v0.x, v0.y, v0.z, v0.w, v1.x, v1.y, v1.z, v1.w,
                      v2.x, v2.y, v2.z, v2.w, v3.x, v3.y, v3.z, v3.w};
    int chunk = ch >> 5, kin = ch & 31;
    __bf16* fb = &A_frag[(n * 16 + chunk) * 512];
    #pragma unroll
    for (int w = 0; w < 16; ++w)
      fb[afrag_idx(w, kin)] = f2bf(vals[w] * rms_s[n * WTILE + w]);
  }
  __syncthreads();

  // ---- dots[w,head,n] = sum_ch A(n,ch,w) * u[ch,head] ----
  {
    int w = t >> 4, n = (t >> 1) & 7, hh = (t & 1) * 4;
    float a0 = 0.f, a1 = 0.f, a2 = 0.f, a3 = 0.f;
    for (int ch = 0; ch < HIDDEN; ++ch) {
      int chunk = ch >> 5, kin = ch & 31;
      float a = bf2f(A_frag[(n * 16 + chunk) * 512 + afrag_idx(w, kin)]);
      float4 uv = *(const float4*)&u_lds[ch * NHEAD + hh];   // one ds_load_b128
      a0 += a * uv.x; a1 += a * uv.y; a2 += a * uv.z; a3 += a * uv.w;
    }
    dots_lds[(w * NHEAD + hh + 0) * NTOK + n] = a0;
    dots_lds[(w * NHEAD + hh + 1) * NTOK + n] = a1;
    dots_lds[(w * NHEAD + hh + 2) * NTOK + n] = a2;
    dots_lds[(w * NHEAD + hh + 3) * NTOK + n] = a3;
  }
  __syncthreads();

  // ---- softmax over n (8) per (w, head) ----
  if (t < WTILE * NHEAD) {
    int w = t >> 3, head = t & 7;
    float* dp = &dots_lds[(w * NHEAD + head) * NTOK];
    float m = dp[0];
    #pragma unroll
    for (int n = 1; n < NTOK; ++n) m = fmaxf(m, dp[n]);
    float e[NTOK], s = 0.f;
    #pragma unroll
    for (int n = 0; n < NTOK; ++n) { e[n] = __expf(dp[n] - m); s += e[n]; }
    float inv = 1.f / s;
    float* ap = &attn_lds[(w * NHEAD + head) * NTOK];
    #pragma unroll
    for (int n = 0; n < NTOK; ++n) ap[n] = e[n] * inv;
  }
  __syncthreads();

  // ---- per head-pair: mixed = sum_n attn*A (vectorized per lane-block),
  //      then o_h = mixed @ gWv[:,head] via WMMA ----
  for (int hp = 0; hp < 4; ++hp) {
    // Wave-level token mix: one (head_sub, chunk) fragment per task.
    // All 16 halfwords of a lane-block share w = lane&15 -> same attn weights.
    #pragma unroll
    for (int i = 0; i < 4; ++i) {
      int task = wave * 4 + i;            // 0..31
      int hsub = task >> 4;               // 0/1
      int chunk = task & 15;
      int head = hp * 2 + hsub;
      int w = lane & 15;
      float4 an0 = *(const float4*)&attn_lds[(w * NHEAD + head) * NTOK + 0];
      float4 an1 = *(const float4*)&attn_lds[(w * NHEAD + head) * NTOK + 4];
      float an[NTOK] = {an0.x, an0.y, an0.z, an0.w, an1.x, an1.y, an1.z, an1.w};
      float accf[16];
      #pragma unroll
      for (int e = 0; e < 16; ++e) accf[e] = 0.f;
      #pragma unroll
      for (int n = 0; n < NTOK; ++n) {
        const uint4* ab = (const uint4*)&A_frag[(n * 16 + chunk) * 512 + lane * 16];
        uint4 d0 = ab[0], d1 = ab[1];
        unsigned dw[8] = {d0.x, d0.y, d0.z, d0.w, d1.x, d1.y, d1.z, d1.w};
        float a_n = an[n];
        #pragma unroll
        for (int j2 = 0; j2 < 8; ++j2) {
          float lo = __builtin_bit_cast(float, dw[j2] << 16);
          float hi = __builtin_bit_cast(float, dw[j2] & 0xFFFF0000u);
          accf[2 * j2 + 0] += a_n * lo;
          accf[2 * j2 + 1] += a_n * hi;
        }
      }
      uint4 p0, p1;
      unsigned pk[8];
      #pragma unroll
      for (int j2 = 0; j2 < 8; ++j2)
        pk[j2] = (rne_bits(accf[2 * j2]) >> 16) |
                 (rne_bits(accf[2 * j2 + 1]) & 0xFFFF0000u);
      p0.x = pk[0]; p0.y = pk[1]; p0.z = pk[2]; p0.w = pk[3];
      p1.x = pk[4]; p1.y = pk[5]; p1.z = pk[6]; p1.w = pk[7];
      uint4* mb = (uint4*)&mixed_frag[(hsub * 16 + chunk) * 512 + lane * 16];
      mb[0] = p0;
      mb[1] = p1;
    }
    __syncthreads();
    // GEMM2: o_h(16x16 per wave) = mixed @ gWv[:, head*64+ct*16 ..]
    // Explicit 2-deep software pipeline: load chunk n+1 before wmma(chunk n).
    {
      int hsub = wave >> 2, ct = wave & 3;
      int head = hp * 2 + hsub;
      const __bf16* abase = &mixed_frag[(hsub * 16) * 512 + lane * 16];
      const __bf16* bbase = wv_frag + ((head * 4 + ct) * 16) * 512 + lane * 16;
      v8f acc = {};
      v16bf a_cur = *(const v16bf*)(abase);
      v16bf b_cur = *(const v16bf*)(bbase);
      #pragma unroll
      for (int chunk = 0; chunk < 16; ++chunk) {
        v16bf a_nxt, b_nxt;
        if (chunk < 15) {
          a_nxt = *(const v16bf*)(abase + (chunk + 1) * 512);
          b_nxt = *(const v16bf*)(bbase + (chunk + 1) * 512);
        }
        if (chunk < 14)
          __builtin_prefetch((const void*)(bbase + (chunk + 2) * 512), 0, 1);
        acc = __builtin_amdgcn_wmma_f32_16x16x32_bf16(false, a_cur, false, b_cur,
                                                      (short)0, acc, false, false);
        a_cur = a_nxt;
        b_cur = b_nxt;
      }
      // scatter D (M=w rows, N=o-col) into A-fragment-ready o_frag (bf16)
      int wl = ((lane >> 4) & 1) * 8;
      int kcol = head * 64 + ct * 16 + (lane & 15);
      int chunk_o = kcol >> 5, kin = kcol & 31;
      #pragma unroll
      for (int r = 0; r < 8; ++r)
        o_frag[chunk_o * 512 + afrag_idx(wl + r, kin)] = f2bf(acc[r]);
    }
    __syncthreads();
  }

  // ---- GEMM3: out tile = o @ Wout + bout. Chunk-outer / 4 col-tiles inner:
  //      one A ds_load feeds 4 WMMAs; A and B double-buffered across chunks.
  {
    const __bf16* abase = &o_frag[lane * 16];
    const __bf16* bb = wout_frag + lane * 16;
    v8f acc0 = {}, acc1 = {}, acc2 = {}, acc3 = {};
    v16bf a_cur = *(const v16bf*)(abase);
    v16bf b0 = *(const v16bf*)(bb + ((wave + 0) * 16 + 0) * 512);
    v16bf b1 = *(const v16bf*)(bb + ((wave + 8) * 16 + 0) * 512);
    v16bf b2 = *(const v16bf*)(bb + ((wave + 16) * 16 + 0) * 512);
    v16bf b3 = *(const v16bf*)(bb + ((wave + 24) * 16 + 0) * 512);
    #pragma unroll
    for (int chunk = 0; chunk < 16; ++chunk) {
      v16bf a_nxt, b0n, b1n, b2n, b3n;
      if (chunk < 15) {
        a_nxt = *(const v16bf*)(abase + (chunk + 1) * 512);
        b0n = *(const v16bf*)(bb + ((wave + 0) * 16 + chunk + 1) * 512);
        b1n = *(const v16bf*)(bb + ((wave + 8) * 16 + chunk + 1) * 512);
        b2n = *(const v16bf*)(bb + ((wave + 16) * 16 + chunk + 1) * 512);
        b3n = *(const v16bf*)(bb + ((wave + 24) * 16 + chunk + 1) * 512);
      }
      acc0 = __builtin_amdgcn_wmma_f32_16x16x32_bf16(false, a_cur, false, b0,
                                                     (short)0, acc0, false, false);
      acc1 = __builtin_amdgcn_wmma_f32_16x16x32_bf16(false, a_cur, false, b1,
                                                     (short)0, acc1, false, false);
      acc2 = __builtin_amdgcn_wmma_f32_16x16x32_bf16(false, a_cur, false, b2,
                                                     (short)0, acc2, false, false);
      acc3 = __builtin_amdgcn_wmma_f32_16x16x32_bf16(false, a_cur, false, b3,
                                                     (short)0, acc3, false, false);
      a_cur = a_nxt;
      b0 = b0n; b1 = b1n; b2 = b2n; b3 = b3n;
    }
    // store 4 col-tiles: lane covers col = ct*16 + lane%16, rows w contiguous
    v8f accs[4] = {acc0, acc1, acc2, acc3};
    #pragma unroll
    for (int j = 0; j < 4; ++j) {
      int ct = wave + 8 * j;
      int col = ct * 16 + (lane & 15);
      float bias = bout[col];
      int wbase = w0 + ((lane >> 4) & 1) * 8;  // M=r (lanes 0-15) / 8+r (16-31)
      float* dst = out + (((b * HIDDEN + col) * 32 + h) * 32 + wbase);
      v8f a = accs[j];
      float4 lo, hi;
      lo.x = a[0] + bias; lo.y = a[1] + bias; lo.z = a[2] + bias; lo.w = a[3] + bias;
      hi.x = a[4] + bias; hi.y = a[5] + bias; hi.z = a[6] + bias; hi.w = a[7] + bias;
      *(float4*)(dst + 0) = lo;
      *(float4*)(dst + 4) = hi;
    }
  }
}

// ---------------------------------------------------------------------------
extern "C" void kernel_launch(void* const* d_in, const int* in_sizes, int n_in,
                              void* d_out, int out_size, void* d_ws, size_t ws_size,
                              hipStream_t stream) {
  const int*   q    = (const int*)d_in[0];
  const float* c    = (const float*)d_in[1];
  const float* emb  = (const float*)d_in[2];
  const float* Wq   = (const float*)d_in[3];
  const float* bq   = (const float*)d_in[4];
  const float* Wkv  = (const float*)d_in[5];
  const float* Wout = (const float*)d_in[6];
  const float* bout = (const float*)d_in[7];
  const float* g    = (const float*)d_in[8];
  float* out = (float*)d_out;

  char* ws = (char*)d_ws;
  float*  u_g       = (float*)(ws + 0);                 // 8*4096*4   = 128 KB
  float*  rms_g     = (float*)(ws + (128 << 10));       // 65536*4    = 256 KB
  __bf16* wv_frag   = (__bf16*)(ws + (384 << 10));      // 512*512*2  = 512 KB
  __bf16* wout_frag = (__bf16*)(ws + (896 << 10));      // 512*512*2  = 512 KB

  qu_kernel<<<8, 256, 0, stream>>>(q, emb, Wq, bq, Wkv, g, u_g);
  repack_kernel<<<dim3(1024, 2), 256, 0, stream>>>(Wkv, Wout, g, wv_frag, wout_frag);
  rms_kernel<<<256, 256, 0, stream>>>(c, rms_g);
  attn_main<<<dim3(2, 32, 8), 256, 0, stream>>>(c, u_g, rms_g, wv_frag, wout_frag,
                                                bout, out);
}
